// Bert_71871982731564
// MI455X (gfx1250) — compile-verified, matrix operand order checked by reading, not run on hardware
//
#include <hip/hip_runtime.h>
#include <hip/hip_bf16.h>
#include <math.h>
#include <stdint.h>

// ---------------- problem constants (match reference) ----------------
#define BB   8
#define SS   512
#define HH   768
#define NHH  12
#define LL   12
#define FF   3072
#define DHH  64
#define MTOK (BB * SS)          // 4096 token rows

typedef __attribute__((ext_vector_type(16))) __bf16 v16bf;
typedef __attribute__((ext_vector_type(8)))  __bf16 bf16x8;
typedef __attribute__((ext_vector_type(8)))  float  v8f;
typedef __attribute__((ext_vector_type(4)))  unsigned int u32x4;
typedef __attribute__((ext_vector_type(8)))  int i32x8;
typedef __attribute__((ext_vector_type(4)))  int i32x4;

#if __has_builtin(__builtin_amdgcn_tensor_load_to_lds) && \
    __has_builtin(__builtin_amdgcn_s_wait_tensorcnt)
#define HAVE_TDM 1
#else
#define HAVE_TDM 0
#endif

// K index of VGPR-pair v for 16-bit WMMA A/B fragments (ISA 7.12.2):
// lanes 0-15 (kh=0): v0..3 -> K=0..7, v4..7 -> K=16..23
// lanes 16-31 (kh=1): v0..3 -> K=8..15, v4..7 -> K=24..31
__device__ __forceinline__ int wmma_kpair(int v, int kh) {
    int base = (v < 4) ? (2 * v) : (16 + 2 * (v - 4));
    return base + kh * 8;
}

// Build a 16-bit A/B fragment when the lane's 16 elements are the two
// contiguous 8-element runs  p[kh*8 .. kh*8+7]  and  p[16+kh*8 .. 16+kh*8+7].
__device__ __forceinline__ v16bf frag_from_contig(const __bf16* p, int kh) {
    bf16x8 lo = *(const bf16x8*)(p + kh * 8);
    bf16x8 hi = *(const bf16x8*)(p + 16 + kh * 8);
    return __builtin_shufflevector(lo, hi, 0, 1, 2, 3, 4, 5, 6, 7, 8, 9, 10,
                                   11, 12, 13, 14, 15);
}

// gfx1250 async global->LDS copy, 16B per lane (ASYNCcnt tracked)
__device__ __forceinline__ void async_copy16(uint32_t lds_off,
                                             const void* gptr) {
    asm volatile("global_load_async_to_lds_b128 %0, %1, off"
                 :
                 : "v"(lds_off), "v"(gptr)
                 : "memory");
}
__device__ __forceinline__ void wait_async0() {
    asm volatile("s_wait_asynccnt 0x0" ::: "memory");
}

#if HAVE_TDM
// One-shot 2D TDM tile load: global (row-major, elem stride0) -> LDS,
// bf16 elements, optional LDS row padding (ISA ch.8 D# layout).
// pad_interval: pad after (1<<pad_interval)*8 bytes; pad_amount: (v+1) DWORDs.
__device__ __forceinline__ void tdm_load_2d(uint32_t lds_addr,
                                            const void* gaddr,
                                            uint32_t tensor_d0,
                                            uint32_t tensor_d1,
                                            uint32_t tile_d0, uint32_t tile_d1,
                                            uint32_t stride0,
                                            uint32_t pad_interval,
                                            uint32_t pad_amount) {
    uint64_t ga = (uint64_t)(uintptr_t)gaddr;
    u32x4 g0;
    g0[0] = 1u;                                   // count=1, user descriptor
    g0[1] = lds_addr;                             // LDS byte address
    g0[2] = (uint32_t)ga;                         // global_addr[31:0]
    g0[3] = ((uint32_t)(ga >> 32) & 0x01FFFFFFu)  // global_addr[56:32]
            | (2u << 30);                         // type = 2 ("image")
    i32x8 g1;
    g1[0] = (int)((1u << 16)                      // data_size = 2 bytes
                  | (1u << 20)                    // pad_enable
                  | (pad_interval << 22) | (pad_amount << 25));
    g1[1] = (int)(tensor_d0 << 16);               // tensor_dim0[15:0]
    g1[2] = (int)((tensor_d0 >> 16) | (tensor_d1 << 16));
    g1[3] = (int)((tensor_d1 >> 16) | (tile_d0 << 16));
    g1[4] = (int)tile_d1;                         // tile_dim1 (tile_dim2=0)
    g1[5] = (int)stride0;                         // tensor_dim0_stride[31:0]
    g1[6] = 0;
    g1[7] = 0;
    i32x4 z4 = {0, 0, 0, 0};                      // 2D: groups 2/3 unused
    i32x8 z8 = {0, 0, 0, 0, 0, 0, 0, 0};
    __builtin_amdgcn_tensor_load_to_lds(g0, g1, z4, z4, z8, 0);
}
#endif

// ---------------- block reduction helper (256 threads) ----------------
__device__ __forceinline__ float block_sum_256(float v, float* red) {
    int tid = threadIdx.x;
    red[tid] = v;
    __syncthreads();
    for (int s = 128; s > 0; s >>= 1) {
        if (tid < s) red[tid] += red[tid + s];
        __syncthreads();
    }
    float r = red[0];
    __syncthreads();
    return r;
}

// 16-lane butterfly reductions (wave32: xor masks 1,2,4,8 stay in half)
__device__ __forceinline__ float rmax16(float x) {
#pragma unroll
    for (int m = 8; m >= 1; m >>= 1) x = fmaxf(x, __shfl_xor(x, m, 32));
    return x;
}
__device__ __forceinline__ float rsum16(float x) {
#pragma unroll
    for (int m = 8; m >= 1; m >>= 1) x += __shfl_xor(x, m, 32);
    return x;
}

// ---------------- fp32 -> bf16 conversion ----------------
__global__ void cvt_f32_bf16_kernel(const float* __restrict__ in,
                                    __bf16* __restrict__ out, int n) {
    int i = blockIdx.x * 256 + threadIdx.x;
    if (i < n) out[i] = (__bf16)in[i];
}

// ---------------- embedding + LayerNorm ----------------
__global__ __launch_bounds__(256) void embed_ln_kernel(
    const int* __restrict__ ids, const int* __restrict__ seg,
    const float* __restrict__ tok, const float* __restrict__ pos,
    const float* __restrict__ typ, const float* __restrict__ g,
    const float* __restrict__ be, float* __restrict__ xo,
    __bf16* __restrict__ xb) {
    __shared__ float red[256];
    int row = blockIdx.x;           // b*S + s
    int s   = row % SS;
    int tid = threadIdx.x;
    int id  = ids[row];
    int sg  = seg[row];
    float v[3];
    float acc = 0.0f;
#pragma unroll
    for (int j = 0; j < 3; ++j) {
        int c = tid + j * 256;
        v[j]  = tok[(size_t)id * HH + c] + typ[(size_t)sg * HH + c] +
                pos[(size_t)s * HH + c];
        acc += v[j];
    }
    float mean = block_sum_256(acc, red) * (1.0f / HH);
    float sq = 0.0f;
#pragma unroll
    for (int j = 0; j < 3; ++j) { float d = v[j] - mean; sq += d * d; }
    float var = block_sum_256(sq, red) * (1.0f / HH);
    float inv = rsqrtf(var + 1e-12f);
    size_t base = (size_t)row * HH;
#pragma unroll
    for (int j = 0; j < 3; ++j) {
        int c = tid + j * 256;
        float y = g[c] * (v[j] - mean) * inv + be[c];
        xo[base + c] = y;
        xb[base + c] = (__bf16)y;
    }
}

// ---------------- bias + residual + LayerNorm ----------------
__global__ __launch_bounds__(256) void bias_res_ln_kernel(
    const float* __restrict__ t, const float* __restrict__ bias,
    const float* __restrict__ res, const float* __restrict__ g,
    const float* __restrict__ bt, float* __restrict__ of,
    __bf16* __restrict__ ob) {
    __shared__ float red[256];
    int row = blockIdx.x;
    int tid = threadIdx.x;
    size_t base = (size_t)row * HH;
    float v[3];
    float acc = 0.0f;
#pragma unroll
    for (int j = 0; j < 3; ++j) {
        int c = tid + j * 256;
        v[j]  = t[base + c] + bias[c] + res[base + c];
        acc += v[j];
    }
    float mean = block_sum_256(acc, red) * (1.0f / HH);
    float sq = 0.0f;
#pragma unroll
    for (int j = 0; j < 3; ++j) { float d = v[j] - mean; sq += d * d; }
    float var = block_sum_256(sq, red) * (1.0f / HH);
    float inv = rsqrtf(var + 1e-12f);
#pragma unroll
    for (int j = 0; j < 3; ++j) {
        int c = tid + j * 256;
        float y = g[c] * (v[j] - mean) * inv + bt[c];
        of[base + c] = y;
        ob[base + c] = (__bf16)y;
    }
}

// ---------------- tiled bf16 WMMA GEMM: C[M,N] = A[M,K] * B[K,N] ----------------
// MODE 0: out = bf16(acc + bias)
// MODE 1: out = f32(acc [+ bias])
// MODE 2: out = bf16(gelu_exact(acc + bias))
#define GBM 128
#define GBN 128
#define GBK 32
template <int MODE, bool HASBIAS>
__global__ __launch_bounds__(256) void gemm_bf16_wmma_kernel(
    const __bf16* __restrict__ A, const __bf16* __restrict__ Bw,
    const float* __restrict__ bias, void* __restrict__ Cout, int M, int N,
    int K) {
    __shared__ __bf16 As[2][GBM][40];    // 80B rows  (pad 16B: interval 3)
    __shared__ __bf16 Bs[2][GBK][136];   // 272B rows (pad 16B: interval 5)
    const int tid  = threadIdx.x;
    const int lane = tid & 31;
    const int wave = tid >> 5;
    const int wm = wave & 3;             // 4 waves along M (32 rows each)
    const int wn = wave >> 2;            // 2 waves along N (64 cols each)
    const int m0 = blockIdx.y * GBM;
    const int n0 = blockIdx.x * GBN;
    const int lr = lane & 15;
    const int kh = lane >> 4;
    const bool lead_wave = (wave == 0);

    // per-thread async-copy coordinates (16B per copy) for the A tile
    const int arow0 = (tid * 8) >> 5;    // 0..63
    const int acol  = (tid * 8) & 31;
#if !HAVE_TDM
    const int brow0 = (tid * 8) >> 7;    // 0..15
    const int bcol  = (tid * 8) & 127;
#endif

    auto issue_tile = [&](int buf, int kt) {
        // A tile (128x32) : per-lane async global->LDS copies
        async_copy16((uint32_t)(uintptr_t)&As[buf][arow0][acol],
                     A + (size_t)(m0 + arow0) * K + kt + acol);
        async_copy16((uint32_t)(uintptr_t)&As[buf][arow0 + 64][acol],
                     A + (size_t)(m0 + arow0 + 64) * K + kt + acol);
        // B tile (32x128) : one TDM descriptor issued by the lead wave
#if HAVE_TDM
        if (lead_wave)
            tdm_load_2d((uint32_t)(uintptr_t)&Bs[buf][0][0],
                        Bw + (size_t)kt * N + n0, (uint32_t)N, (uint32_t)K,
                        GBN, GBK, (uint32_t)N, /*pad_interval=*/5,
                        /*pad_amount=*/3);
#else
        async_copy16((uint32_t)(uintptr_t)&Bs[buf][brow0][bcol],
                     Bw + (size_t)(kt + brow0) * N + n0 + bcol);
        async_copy16((uint32_t)(uintptr_t)&Bs[buf][brow0 + 16][bcol],
                     Bw + (size_t)(kt + brow0 + 16) * N + n0 + bcol);
#endif
    };
    auto wait_tiles = [&]() {
        wait_async0();
#if HAVE_TDM
        if (lead_wave) __builtin_amdgcn_s_wait_tensorcnt(0);
#endif
    };

    v8f acc[2][4] = {};

    issue_tile(0, 0);
    wait_tiles();
    __syncthreads();

    int buf = 0;
    for (int kt = 0; kt < K; kt += GBK) {
        if (kt + GBK < K) issue_tile(buf ^ 1, kt + GBK);

        v16bf af[2], bfr[4];
#pragma unroll
        for (int mi = 0; mi < 2; ++mi) {
            int row = wm * 32 + mi * 16 + lr;
            af[mi]  = frag_from_contig(&As[buf][row][0], kh);
        }
#pragma unroll
        for (int ni = 0; ni < 4; ++ni) {
            int col = wn * 64 + ni * 16 + lr;
#pragma unroll
            for (int v = 0; v < 8; ++v) {
                int k           = wmma_kpair(v, kh);
                bfr[ni][2 * v]     = Bs[buf][k][col];
                bfr[ni][2 * v + 1] = Bs[buf][k + 1][col];
            }
        }
#pragma unroll
        for (int mi = 0; mi < 2; ++mi)
#pragma unroll
            for (int ni = 0; ni < 4; ++ni)
                acc[mi][ni] = __builtin_amdgcn_wmma_f32_16x16x32_bf16(
                    false, af[mi], false, bfr[ni], (short)0, acc[mi][ni],
                    false, false);

        wait_tiles();
        __syncthreads();
        buf ^= 1;
    }

    // epilogue (C layout: lane holds rows (kh*8 + r), col = lr per 16x16 tile)
#pragma unroll
    for (int ni = 0; ni < 4; ++ni) {
        int colg = n0 + wn * 64 + ni * 16 + lr;
        float bv = HASBIAS ? bias[colg] : 0.0f;
#pragma unroll
        for (int mi = 0; mi < 2; ++mi) {
#pragma unroll
            for (int r = 0; r < 8; ++r) {
                int rowg  = m0 + wm * 32 + mi * 16 + kh * 8 + r;
                float val = acc[mi][ni][r] + bv;
                size_t idx = (size_t)rowg * N + colg;
                if (MODE == 0) {
                    ((__bf16*)Cout)[idx] = (__bf16)val;
                } else if (MODE == 1) {
                    ((float*)Cout)[idx] = val;
                } else {
                    float gl = 0.5f * val * (1.0f + erff(val * 0.70710678118f));
                    ((__bf16*)Cout)[idx] = (__bf16)gl;
                }
            }
        }
    }
}

// ---------------- flash-style attention: one wave per (b, h, 16 queries) ----------------
__global__ __launch_bounds__(32) void attn_kernel(
    const __bf16* __restrict__ Q, const __bf16* __restrict__ Km,
    const __bf16* __restrict__ V, const float* __restrict__ mask,
    __bf16* __restrict__ Ctx) {
    __shared__ __bf16 Ps[16][40];
    const int lane = threadIdx.x;
    const int lr   = lane & 15;
    const int kh   = lane >> 4;
    const int s0   = blockIdx.x * 16;
    const int h    = blockIdx.y;
    const int b    = blockIdx.z;
    const float scale = 0.125f;      // 1/sqrt(64)

    // Q fragment: A matrix [16 queries x 64 dh] -> 2 k-chunks of 32
    const __bf16* qrow = Q + ((size_t)b * SS + s0 + lr) * HH + (size_t)h * DHH;
    v16bf aq[2];
#pragma unroll
    for (int c = 0; c < 2; ++c) aq[c] = frag_from_contig(qrow + c * 32, kh);

    float mrow[8], lrow[8];
#pragma unroll
    for (int r = 0; r < 8; ++r) { mrow[r] = -3.0e38f; lrow[r] = 0.0f; }
    v8f o[4] = {};

    for (int kb = 0; kb < SS / 32; ++kb) {
        // ---- scores: 2 tiles of 16x16 over key block of 32 ----
        v8f ct[2] = {};
#pragma unroll
        for (int t = 0; t < 2; ++t) {
            int key = kb * 32 + t * 16 + lr;     // B col = key
            const __bf16* krow =
                Km + ((size_t)b * SS + key) * HH + (size_t)h * DHH;
#pragma unroll
            for (int c = 0; c < 2; ++c) {
                v16bf bk = frag_from_contig(krow + c * 32, kh);
                ct[t] = __builtin_amdgcn_wmma_f32_16x16x32_bf16(
                    false, aq[c], false, bk, (short)0, ct[t], false, false);
            }
        }
        float mk0 = (1.0f - mask[(size_t)b * SS + kb * 32 + lr])      * -10000.0f;
        float mk1 = (1.0f - mask[(size_t)b * SS + kb * 32 + 16 + lr]) * -10000.0f;

        // ---- online softmax (each lane owns rows kh*8 + r) ----
#pragma unroll
        for (int r = 0; r < 8; ++r) {
            float sv0 = ct[0][r] * scale + mk0;
            float sv1 = ct[1][r] * scale + mk1;
            float mx   = rmax16(fmaxf(sv0, sv1));
            float mnew = fmaxf(mrow[r], mx);
            float p0 = expf(sv0 - mnew);
            float p1 = expf(sv1 - mnew);
            float rs = rsum16(p0 + p1);
            float alpha = expf(mrow[r] - mnew);
            lrow[r] = lrow[r] * alpha + rs;
            mrow[r] = mnew;
#pragma unroll
            for (int t = 0; t < 4; ++t) o[t][r] *= alpha;
            Ps[kh * 8 + r][lr]      = (__bf16)p0;
            Ps[kh * 8 + r][16 + lr] = (__bf16)p1;
        }
        __syncthreads();

        // ---- P (16x32) as A fragment ----
        v16bf ap = frag_from_contig(&Ps[lr][0], kh);

        // ---- o += P @ V : 4 dh tiles ----
#pragma unroll
        for (int t = 0; t < 4; ++t) {
            int dh = t * 16 + lr;                 // B col = dh
            v16bf bv;
#pragma unroll
            for (int v = 0; v < 8; ++v) {
                int key       = kb * 32 + wmma_kpair(v, kh);
                bv[2 * v]     = V[((size_t)b * SS + key) * HH + (size_t)h * DHH + dh];
                bv[2 * v + 1] = V[((size_t)b * SS + key + 1) * HH + (size_t)h * DHH + dh];
            }
            o[t] = __builtin_amdgcn_wmma_f32_16x16x32_bf16(
                false, ap, false, bv, (short)0, o[t], false, false);
        }
        __syncthreads();
    }

    // ---- epilogue: ctx = o / l ----
#pragma unroll
    for (int t = 0; t < 4; ++t) {
        int dh = t * 16 + lr;
#pragma unroll
        for (int r = 0; r < 8; ++r) {
            int row = s0 + kh * 8 + r;
            Ctx[((size_t)b * SS + row) * HH + (size_t)h * DHH + dh] =
                (__bf16)(o[t][r] / lrow[r]);
        }
    }
}

// ---------------- host-side orchestration ----------------
extern "C" void kernel_launch(void* const* d_in, const int* in_sizes, int n_in,
                              void* d_out, int out_size, void* d_ws,
                              size_t ws_size, hipStream_t stream) {
    (void)in_sizes; (void)n_in; (void)out_size; (void)ws_size;

    const int*   ids   = (const int*)d_in[0];
    const int*   seg   = (const int*)d_in[1];
    const float* amask = (const float*)d_in[2];
    const float* tok   = (const float*)d_in[3];
    const float* pos   = (const float*)d_in[4];
    const float* typ   = (const float*)d_in[5];
    const float* eg    = (const float*)d_in[6];
    const float* eb    = (const float*)d_in[7];
    const float* Wq    = (const float*)d_in[8];
    const float* bq    = (const float*)d_in[9];
    const float* Wk    = (const float*)d_in[10];
    const float* bk    = (const float*)d_in[11];
    const float* Wv    = (const float*)d_in[12];
    const float* bv    = (const float*)d_in[13];
    const float* Wao   = (const float*)d_in[14];
    const float* bao   = (const float*)d_in[15];
    const float* ln1g  = (const float*)d_in[16];
    const float* ln1b  = (const float*)d_in[17];
    const float* Wi    = (const float*)d_in[18];
    const float* bi    = (const float*)d_in[19];
    const float* Wio   = (const float*)d_in[20];
    const float* bio   = (const float*)d_in[21];
    const float* ln2g  = (const float*)d_in[22];
    const float* ln2b  = (const float*)d_in[23];

    // workspace carve-out
    char* wp = (char*)d_ws;
    auto carve = [&](size_t bytes) {
        void* p = (void*)wp;
        wp += (bytes + 255) & ~(size_t)255;
        return p;
    };
    float*  x      = (float*)carve((size_t)MTOK * HH * 4);
    __bf16* xbf    = (__bf16*)carve((size_t)MTOK * HH * 2);
    __bf16* qbf    = (__bf16*)carve((size_t)MTOK * HH * 2);
    __bf16* kbf    = (__bf16*)carve((size_t)MTOK * HH * 2);
    __bf16* vbf    = (__bf16*)carve((size_t)MTOK * HH * 2);
    __bf16* ctxbf  = (__bf16*)carve((size_t)MTOK * HH * 2);
    float*  attnf  = (float*)carve((size_t)MTOK * HH * 4);
    __bf16* attnbf = (__bf16*)carve((size_t)MTOK * HH * 2);
    __bf16* hbf    = (__bf16*)carve((size_t)MTOK * FF * 2);
    float*  tmpf   = (float*)carve((size_t)MTOK * HH * 4);
    __bf16* wbuf   = (__bf16*)carve((size_t)HH * FF * 2);

    const int nHH = HH * HH;
    const int nHF = HH * FF;

    // embedding + LN
    embed_ln_kernel<<<MTOK, 256, 0, stream>>>(ids, seg, tok, pos, typ, eg, eb,
                                              x, xbf);

    dim3 gH(HH / GBN, MTOK / GBM);   // N=768  -> 6 x 32
    dim3 gF(FF / GBN, MTOK / GBM);   // N=3072 -> 24 x 32
    dim3 attnGrid(SS / 16, NHH, BB);

    for (int l = 0; l < LL; ++l) {
        // Q
        cvt_f32_bf16_kernel<<<(nHH + 255) / 256, 256, 0, stream>>>(
            Wq + (size_t)l * nHH, wbuf, nHH);
        gemm_bf16_wmma_kernel<0, true><<<gH, 256, 0, stream>>>(
            xbf, wbuf, bq + (size_t)l * HH, qbf, MTOK, HH, HH);
        // K
        cvt_f32_bf16_kernel<<<(nHH + 255) / 256, 256, 0, stream>>>(
            Wk + (size_t)l * nHH, wbuf, nHH);
        gemm_bf16_wmma_kernel<0, true><<<gH, 256, 0, stream>>>(
            xbf, wbuf, bk + (size_t)l * HH, kbf, MTOK, HH, HH);
        // V
        cvt_f32_bf16_kernel<<<(nHH + 255) / 256, 256, 0, stream>>>(
            Wv + (size_t)l * nHH, wbuf, nHH);
        gemm_bf16_wmma_kernel<0, true><<<gH, 256, 0, stream>>>(
            xbf, wbuf, bv + (size_t)l * HH, vbf, MTOK, HH, HH);
        // attention
        attn_kernel<<<attnGrid, 32, 0, stream>>>(qbf, kbf, vbf, amask, ctxbf);
        // ctx @ Wao -> tmp f32
        cvt_f32_bf16_kernel<<<(nHH + 255) / 256, 256, 0, stream>>>(
            Wao + (size_t)l * nHH, wbuf, nHH);
        gemm_bf16_wmma_kernel<1, false><<<gH, 256, 0, stream>>>(
            ctxbf, wbuf, nullptr, tmpf, MTOK, HH, HH);
        // attn_out = LN(tmp + bao + x)
        bias_res_ln_kernel<<<MTOK, 256, 0, stream>>>(
            tmpf, bao + (size_t)l * HH, x, ln1g + (size_t)l * HH,
            ln1b + (size_t)l * HH, attnf, attnbf);
        // h = gelu(attn_out @ Wi + bi) -> bf16
        cvt_f32_bf16_kernel<<<(nHF + 255) / 256, 256, 0, stream>>>(
            Wi + (size_t)l * nHF, wbuf, nHF);
        gemm_bf16_wmma_kernel<2, true><<<gF, 256, 0, stream>>>(
            attnbf, wbuf, bi + (size_t)l * FF, hbf, MTOK, FF, HH);
        // h @ Wio -> tmp f32
        cvt_f32_bf16_kernel<<<(nHF + 255) / 256, 256, 0, stream>>>(
            Wio + (size_t)l * nHF, wbuf, nHF);
        gemm_bf16_wmma_kernel<1, false><<<gH, 256, 0, stream>>>(
            hbf, wbuf, nullptr, tmpf, MTOK, HH, FF);
        // x = LN(tmp + bio + attn_out); last layer writes d_out
        float* xdst = (l == LL - 1) ? (float*)d_out : x;
        bias_res_ln_kernel<<<MTOK, 256, 0, stream>>>(
            tmpf, bio + (size_t)l * HH, attnf, ln2g + (size_t)l * HH,
            ln2b + (size_t)l * HH, xdst, xbf);
    }
}
